// MMDoubleStreamBlock_48361331753313
// MI455X (gfx1250) — matrix operand order repaired and checked by reading, not compiled, not run on hardware
//
#include <hip/hip_runtime.h>
#include <cstdint>
#include <cstddef>

// ---------------------------------------------------------------------------
// Problem constants (from reference): B=2, L=2048, C=2048, H=16, D=128
// ---------------------------------------------------------------------------
#define BQ 2
#define LQ 2048
#define CQ 2048
#define HQ 16
#define DQ 128
#define C6 (6 * CQ)
#define CM 8192
#define MQ (BQ * LQ)

typedef __bf16 v16bf __attribute__((ext_vector_type(16)));
typedef float  v8f   __attribute__((ext_vector_type(8)));
typedef unsigned int u32x4 __attribute__((ext_vector_type(4)));
typedef int          i32x4 __attribute__((ext_vector_type(4)));
typedef int          i32x8 __attribute__((ext_vector_type(8)));

struct FragU { unsigned int u[8]; };   // 32 bytes == one WMMA bf16 A/B operand

static __device__ __forceinline__ v8f wmma_bf16(const FragU& a, const FragU& b, v8f c) {
  return __builtin_amdgcn_wmma_f32_16x16x32_bf16(
      false, __builtin_bit_cast(v16bf, a),
      false, __builtin_bit_cast(v16bf, b),
      (short)0, c, false, false);
}

static __device__ __forceinline__ uint16_t f2bf(float f) {  // RNE f32 -> bf16
  unsigned int u = __builtin_bit_cast(unsigned int, f);
  u += 0x7fffu + ((u >> 16) & 1u);
  return (uint16_t)(u >> 16);
}

static __device__ __forceinline__ v8f v8f_zero() {
  v8f z = {0.f, 0.f, 0.f, 0.f, 0.f, 0.f, 0.f, 0.f};
  return z;
}

// ---------------------------------------------------------------------------
// TDM: DMA one 128-row x 32-col bf16 tile from global [rows x K] into LDS.
// LDS rows padded: 16 DWORDs data + 4 DWORDs pad  =>  row stride 40 u16 (LDT).
// D# per CDNA5 ISA ch.8: group0 = {flags, lds_addr, global_addr, type=2},
// group1 = {data_size=2B | pad cfg, tensor_dim0/1, tile_dim0=32, tile_dim1=128,
//           tensor_dim0_stride=K}. Groups 2/3 unused (2-D tensor).
// ---------------------------------------------------------------------------
static __device__ __forceinline__ void tdm_load_tile_128x32(
    const uint16_t* gsrc, unsigned int lds_byte_addr, int K, int rowsAvail) {
  const uint64_t ga = (uint64_t)(uintptr_t)gsrc;
  u32x4 g0;
  g0[0] = 1u;                                                  // count=1 (valid), user mode
  g0[1] = lds_byte_addr;                                       // LDS destination (bytes)
  g0[2] = (unsigned int)(ga & 0xffffffffu);                    // global addr [31:0]
  g0[3] = (unsigned int)((ga >> 32) & 0x01ffffffu) | 0x80000000u;  // addr[56:32] | type=2

  i32x8 g1;
  g1[0] = (1 << 16)        // data_size = 2 bytes
        | (1 << 20)        // pad_enable
        | (3 << 22)        // pad_interval: 16 DWORDs between pads
        | (3 << 25);       // pad_amount:   4 DWORDs  -> LDS row stride 40 u16
  g1[1] = (int)(((unsigned)K & 0xffffu) << 16);                // tensor_dim0[15:0]
  g1[2] = (int)((((unsigned)K >> 16) & 0xffffu)                // tensor_dim0[31:16]
        | (((unsigned)rowsAvail & 0xffffu) << 16));            // tensor_dim1[15:0]
  g1[3] = (int)((((unsigned)rowsAvail >> 16) & 0xffffu)        // tensor_dim1[31:16]
        | (32u << 16));                                        // tile_dim0 = 32
  g1[4] = 128;                                                 // tile_dim1 = 128 (tile_dim2=0)
  g1[5] = K;                                                   // tensor_dim0_stride[31:0]
  g1[6] = 0;                                                   // stride hi | dim1_stride lo
  g1[7] = 0;

  i32x4 gz = {0, 0, 0, 0};
#if defined(__clang_major__) && (__clang_major__ >= 23)
  i32x8 gz8 = {0, 0, 0, 0, 0, 0, 0, 0};
  __builtin_amdgcn_tensor_load_to_lds(g0, g1, gz, gz, gz8, 0);
#else
  __builtin_amdgcn_tensor_load_to_lds(g0, g1, gz, gz, 0);
#endif
}

// ---------------------------------------------------------------------------
// Weight transpose + convert: in [K,N] f32 row-major -> out [N,K] bf16
// ---------------------------------------------------------------------------
__global__ __launch_bounds__(256) void transpose_f32_bf16(
    const float* __restrict__ in, uint16_t* __restrict__ out, int K, int N) {
  __shared__ float t[32][33];
  const int n0 = blockIdx.x * 32, k0 = blockIdx.y * 32;
  const int tx = threadIdx.x, ty = threadIdx.y;  // 32 x 8
#pragma unroll
  for (int r = 0; r < 4; ++r)
    t[ty + r * 8][tx] = in[(size_t)(k0 + ty + r * 8) * N + n0 + tx];
  __syncthreads();
#pragma unroll
  for (int r = 0; r < 4; ++r)
    out[(size_t)(n0 + ty + r * 8) * K + k0 + tx] = f2bf(t[tx][ty + r * 8]);
}

// ---------------------------------------------------------------------------
// mod = silu(vec) @ mod_w + mod_b     (B x 6C, skinny GEMV, VALU)
// ---------------------------------------------------------------------------
__global__ __launch_bounds__(256) void mod_gemv(
    const float* __restrict__ vec, const float* __restrict__ w,
    const float* __restrict__ bias, float* __restrict__ mod) {
  __shared__ float sv[CQ];
  const int b = blockIdx.y;
  const int j = blockIdx.x * 256 + threadIdx.x;
  for (int i = threadIdx.x; i < CQ; i += 256) {
    float x = vec[b * CQ + i];
    sv[i] = x / (1.0f + __expf(-x));
  }
  __syncthreads();
  float acc = bias[j];
  for (int c = 0; c < CQ; ++c) acc += sv[c] * w[(size_t)c * C6 + j];
  mod[(size_t)b * C6 + j] = acc;
}

// ---------------------------------------------------------------------------
// LayerNorm (no affine) + modulate: out = bf16(LN(x)*(1+sc)+sh)
// ---------------------------------------------------------------------------
__global__ __launch_bounds__(256) void ln_modulate(
    const float* __restrict__ x, const float* __restrict__ mod,
    int shOff, int scOff, uint16_t* __restrict__ out) {
  __shared__ float r1[256], r2[256];
  const int row = blockIdx.x;
  const int b = row >> 11;  // L = 2048
  const float* xr = x + (size_t)row * CQ;
  const float* sh = mod + (size_t)b * C6 + shOff;
  const float* sc = mod + (size_t)b * C6 + scOff;
  float v[8];
  float s = 0.f, ss = 0.f;
#pragma unroll
  for (int i = 0; i < 8; ++i) {
    v[i] = xr[threadIdx.x + i * 256];
    s += v[i];
    ss += v[i] * v[i];
  }
  r1[threadIdx.x] = s;
  r2[threadIdx.x] = ss;
  __syncthreads();
  for (int o = 128; o > 0; o >>= 1) {
    if (threadIdx.x < (unsigned)o) {
      r1[threadIdx.x] += r1[threadIdx.x + o];
      r2[threadIdx.x] += r2[threadIdx.x + o];
    }
    __syncthreads();
  }
  const float mu = r1[0] * (1.0f / CQ);
  const float var = r2[0] * (1.0f / CQ) - mu * mu;
  const float rstd = rsqrtf(var + 1e-6f);
#pragma unroll
  for (int i = 0; i < 8; ++i) {
    int c = threadIdx.x + i * 256;
    out[(size_t)row * CQ + c] = f2bf((v[i] - mu) * rstd * (1.0f + sc[c]) + sh[c]);
  }
}

// ---------------------------------------------------------------------------
// WMMA GEMM with TDM-staged, double-buffered LDS tiles.
// C[M,N] = A[M,K](bf16) x W, Bt[N,K] = W^T (bf16). 128x128x32 block tiles,
// 256 threads = 8 waves (4 M x 2 N), 8 WMMA accumulators per wave.
// ---------------------------------------------------------------------------
#define LDT 40  // u16 row stride of TDM-padded LDS tiles

enum { EPI_RMS = 0, EPI_VT = 1, EPI_PROJ = 2, EPI_GELU = 3, EPI_FINAL = 4 };

static __device__ __forceinline__ FragU frag_a_lds(const uint16_t* tile, int row0, int lane) {
  // A 16x32 bf16: lane holds row M=lane&15; lanes 0-15: K {0..7,16..23}, lanes 16-31: +8
  FragU f;
  const uint16_t* p = tile + (row0 + (lane & 15)) * LDT;
  const int kb = (lane >> 4) << 3;
#pragma unroll
  for (int v = 0; v < 8; ++v) {
    int k = kb + ((v >> 2) << 4) + ((v & 3) << 1);
    f.u[v] = *(const unsigned int*)(p + k);
  }
  return f;
}

static __device__ __forceinline__ FragU frag_b_lds(const uint16_t* tile, int row0, int lane) {
  // B 32x16 bf16: lane holds col N=lane&15; lanes 0-15: K {0..7,16..23}, lanes 16-31: +8
  FragU f;
  const uint16_t* p = tile + (row0 + (lane & 15)) * LDT + ((lane >> 4) << 3);
#pragma unroll
  for (int v = 0; v < 8; ++v) f.u[v] = *(const unsigned int*)(p + ((v >> 2) << 4) + ((v & 3) << 1));
  return f;
}

template <int EPI>
__global__ __launch_bounds__(256) void gemm_wmma(
    const uint16_t* __restrict__ A,   // [M x K] bf16
    const uint16_t* __restrict__ Bt,  // [N x K] bf16 (transposed weight)
    int M, int N, int K,
    void* __restrict__ outp,
    const float* __restrict__ bias,   // [N] (GELU/FINAL)
    const float* __restrict__ resid,  // [M x N] f32 (PROJ/FINAL)
    const float* __restrict__ gate,   // mod + chunk*C, index [b*gateStride + n]
    int gateStride,
    const float* __restrict__ rmsw,   // [128] (RMS)
    int L, int H) {
  __shared__ __align__(16) uint16_t sA[2][128 * LDT];
  __shared__ __align__(16) uint16_t sB[2][128 * LDT];
  const int tid = threadIdx.x, lane = tid & 31, wave = tid >> 5;
  const int wm = wave & 3, wn = wave >> 2;
  const int bm = blockIdx.y * 128, bn = blockIdx.x * 128;
  const int halfk8 = (lane >> 4) << 3;
  const bool issuer = (wave == 0);

  v8f acc[2][4];
#pragma unroll
  for (int i = 0; i < 2; ++i)
#pragma unroll
    for (int j = 0; j < 4; ++j) acc[i][j] = v8f_zero();

  const uint16_t* gA = A + (size_t)bm * K;
  const uint16_t* gBt = Bt + (size_t)bn * K;

  // prologue: DMA first tile pair into buffer 0
  if (issuer) {
    tdm_load_tile_128x32(gA, (unsigned int)(uintptr_t)&sA[0][0], K, M - bm);
    tdm_load_tile_128x32(gBt, (unsigned int)(uintptr_t)&sB[0][0], K, N - bn);
  }

  int cur = 0;
  for (int k0 = 0; k0 < K; k0 += 32) {
    if (issuer) __builtin_amdgcn_s_wait_tensorcnt(0);  // buffer `cur` complete
    __syncthreads();  // release compute waves; orders prior reads of cur^1 before reuse
    if (issuer && (k0 + 32 < K)) {  // DMA next tile pair while we compute
      tdm_load_tile_128x32(gA + (k0 + 32), (unsigned int)(uintptr_t)&sA[cur ^ 1][0],
                           K - (k0 + 32), M - bm);
      tdm_load_tile_128x32(gBt + (k0 + 32), (unsigned int)(uintptr_t)&sB[cur ^ 1][0],
                           K - (k0 + 32), N - bn);
    }
    if (k0 + 64 < K) {  // long-range prefetch of the slab after next
      __builtin_prefetch(gA + (size_t)(tid >> 1) * K + k0 + 64, 0, 3);
      __builtin_prefetch(gBt + (size_t)(tid >> 1) * K + k0 + 64, 0, 3);
    }
    FragU a0 = frag_a_lds(sA[cur], wm * 32, lane);
    FragU a1 = frag_a_lds(sA[cur], wm * 32 + 16, lane);
#pragma unroll
    for (int j = 0; j < 4; ++j) {
      FragU bf = frag_b_lds(sB[cur], wn * 64 + j * 16, lane);
      acc[0][j] = wmma_bf16(a0, bf, acc[0][j]);
      acc[1][j] = wmma_bf16(a1, bf, acc[1][j]);
    }
    cur ^= 1;
  }
  __syncthreads();

  if constexpr (EPI == EPI_RMS) {
    // per-head RMSNorm over the 128-wide N tile (bn is head-aligned)
    __shared__ float ssb[8][32];
    float ssr[2][8];
#pragma unroll
    for (int mi = 0; mi < 2; ++mi)
#pragma unroll
      for (int v = 0; v < 8; ++v) {
        float p = 0.f;
#pragma unroll
        for (int ni = 0; ni < 4; ++ni) p += acc[mi][ni][v] * acc[mi][ni][v];
        p += __shfl_xor(p, 1);
        p += __shfl_xor(p, 2);
        p += __shfl_xor(p, 4);
        p += __shfl_xor(p, 8);
        ssr[mi][v] = p;  // sum over this wave's 64 columns, replicated in half
      }
    if ((lane & 15) == 0) {
#pragma unroll
      for (int mi = 0; mi < 2; ++mi)
#pragma unroll
        for (int v = 0; v < 8; ++v) ssb[wave][mi * 16 + v + halfk8] = ssr[mi][v];
    }
    __syncthreads();
    const int partner = ((1 - wn) << 2) + wm;
    uint16_t* out = (uint16_t*)outp;
#pragma unroll
    for (int mi = 0; mi < 2; ++mi)
#pragma unroll
      for (int v = 0; v < 8; ++v) {
        float total = ssr[mi][v] + ssb[partner][mi * 16 + v + halfk8];
        float rstd = rsqrtf(total * (1.0f / 128.0f) + 1e-6f);
        int r = bm + wm * 32 + mi * 16 + v + halfk8;
#pragma unroll
        for (int ni = 0; ni < 4; ++ni) {
          int c = wn * 64 + ni * 16 + (lane & 15);  // 0..127 == head-dim index
          out[(size_t)r * N + bn + c] = f2bf(acc[mi][ni][v] * rstd * rmsw[c]);
        }
      }
  } else {
#pragma unroll
    for (int mi = 0; mi < 2; ++mi)
#pragma unroll
      for (int ni = 0; ni < 4; ++ni)
#pragma unroll
        for (int v = 0; v < 8; ++v) {
          int r = bm + wm * 32 + mi * 16 + v + halfk8;
          int c = bn + wn * 64 + ni * 16 + (lane & 15);
          float x = acc[mi][ni][v];
          size_t idx = (size_t)r * N + c;
          if constexpr (EPI == EPI_VT) {
            int b = r / L, l = r - b * L;
            int h = c >> 7, d = c & 127;
            ((uint16_t*)outp)[((size_t)(b * H + h) * 128 + d) * L + l] = f2bf(x);
          } else if constexpr (EPI == EPI_PROJ) {
            float g = gate[(size_t)(r / L) * gateStride + c];
            ((float*)outp)[idx] = resid[idx] + g * x;
          } else if constexpr (EPI == EPI_GELU) {
            float t = x + bias[c];
            float gl = 0.5f * t *
                       (1.0f + tanhf(0.7978845608028654f * (t + 0.044715f * t * t * t)));
            ((uint16_t*)outp)[idx] = f2bf(gl);
          } else {  // EPI_FINAL
            float g = gate[(size_t)(r / L) * gateStride + c];
            ((float*)outp)[idx] = resid[idx] + g * (x + bias[c]);
          }
        }
  }
}

// ---------------------------------------------------------------------------
// Flash attention (causal), wave32 WMMA. One wave per 16-row Q tile.
// grid = (B*H, L/128), block = 256 (8 waves). 32-key blocks:
//   S(16x32) = Q(16x128)Kt  : 8 wmma ; O(16x128) += P(16x32)V : 8 wmma.
// ---------------------------------------------------------------------------
__global__ __launch_bounds__(256) void attn_flash(
    const uint16_t* __restrict__ Q, const uint16_t* __restrict__ Kb,
    const uint16_t* __restrict__ Vt, uint16_t* __restrict__ O) {
  __shared__ uint16_t pb[8][16 * 36];  // per-wave P bounce (16x32 bf16, stride 36)
  const int lane = threadIdx.x & 31, wave = threadIdx.x >> 5;
  const int bh = blockIdx.x;
  const int b = bh >> 4;
  const int h = bh & 15;
  const int q0 = (blockIdx.y * 8 + wave) * 16;
  const size_t rowBase = (size_t)b * LQ;
  const int col0 = h * DQ;
  const int halfk8 = (lane >> 4) << 3;

  FragU aq[4];  // Q tile as 4 A-fragments (K = head dim, 4 chunks of 32)
  {
    const uint16_t* qr = Q + (rowBase + q0 + (lane & 15)) * CQ + col0;
#pragma unroll
    for (int c = 0; c < 4; ++c)
#pragma unroll
      for (int v = 0; v < 8; ++v) {
        int k = c * 32 + halfk8 + ((v >> 2) << 4) + ((v & 3) << 1);
        aq[c].u[v] = *(const unsigned int*)(qr + k);
      }
  }

  float m_i[8], l_i[8];
  v8f o[8];
#pragma unroll
  for (int v = 0; v < 8; ++v) { m_i[v] = -3.0e38f; l_i[v] = 0.f; }
#pragma unroll
  for (int t = 0; t < 8; ++t) o[t] = v8f_zero();

  const float scale = 0.08838834764831845f;  // 1/sqrt(128)
  const int nj = (q0 + 16 + 31) >> 5;        // causal: only key blocks <= diagonal
  for (int j = 0; j < nj; ++j) {
    const int kb = j * 32;
    v8f s[2] = {v8f_zero(), v8f_zero()};
#pragma unroll
    for (int sub = 0; sub < 2; ++sub) {
      const uint16_t* kr = Kb + (rowBase + kb + sub * 16 + (lane & 15)) * CQ + col0;
#pragma unroll
      for (int c = 0; c < 4; ++c) {
        FragU kf;  // B-fragment: lane=key column, K pairs = head-dim
#pragma unroll
        for (int v = 0; v < 8; ++v)
          kf.u[v] = *(const unsigned int*)(kr + c * 32 + ((lane >> 4) << 3) + ((v >> 2) << 4) + ((v & 3) << 1));
        s[sub] = wmma_bf16(aq[c], kf, s[sub]);
      }
    }
    // scale, causal mask, online softmax (rows replicated across 16-lane halves)
    float p0[8], p1[8], alpha[8];
#pragma unroll
    for (int v = 0; v < 8; ++v) {
      int r = q0 + v + halfk8;
      float s0 = s[0][v] * scale, s1 = s[1][v] * scale;
      if (kb + (lane & 15) > r) s0 = -3.0e38f;
      if (kb + 16 + (lane & 15) > r) s1 = -3.0e38f;
      float t = fmaxf(s0, s1);
      t = fmaxf(t, __shfl_xor(t, 1));
      t = fmaxf(t, __shfl_xor(t, 2));
      t = fmaxf(t, __shfl_xor(t, 4));
      t = fmaxf(t, __shfl_xor(t, 8));
      float mn = fmaxf(m_i[v], t);
      alpha[v] = __expf(m_i[v] - mn);
      m_i[v] = mn;
      p0[v] = __expf(s0 - mn);
      p1[v] = __expf(s1 - mn);
      float rs = p0[v] + p1[v];
      rs += __shfl_xor(rs, 1);
      rs += __shfl_xor(rs, 2);
      rs += __shfl_xor(rs, 4);
      rs += __shfl_xor(rs, 8);
      l_i[v] = l_i[v] * alpha[v] + rs;
    }
#pragma unroll
    for (int t = 0; t < 8; ++t)
#pragma unroll
      for (int v = 0; v < 8; ++v) o[t][v] *= alpha[v];
    // P (C-layout f32) -> A-fragment (bf16) via wave-private LDS bounce
    uint16_t* pl = pb[wave];
#pragma unroll
    for (int v = 0; v < 8; ++v) {
      int rr = v + halfk8;
      pl[rr * 36 + (lane & 15)] = f2bf(p0[v]);
      pl[rr * 36 + 16 + (lane & 15)] = f2bf(p1[v]);
    }
    FragU pf;
    {
      const uint16_t* pr = pl + (lane & 15) * 36;
#pragma unroll
      for (int v = 0; v < 8; ++v)
        pf.u[v] = *(const unsigned int*)(pr + halfk8 + ((v >> 2) << 4) + ((v & 3) << 1));
    }
#pragma unroll
    for (int t = 0; t < 8; ++t) {
      FragU vf;  // B-fragment from Vt: lane = d column, K pairs = consecutive keys
      const uint16_t* vr = Vt + ((size_t)bh * DQ + t * 16 + (lane & 15)) * LQ + kb;
#pragma unroll
      for (int v = 0; v < 8; ++v)
        vf.u[v] = *(const unsigned int*)(vr + ((lane >> 4) << 3) + ((v >> 2) << 4) + ((v & 3) << 1));
      o[t] = wmma_bf16(pf, vf, o[t]);
    }
  }
#pragma unroll
  for (int t = 0; t < 8; ++t)
#pragma unroll
    for (int v = 0; v < 8; ++v) {
      int r = q0 + v + halfk8;
      int d = t * 16 + (lane & 15);
      O[(rowBase + r) * CQ + col0 + d] = f2bf(o[t][v] / l_i[v]);
    }
}

// ---------------------------------------------------------------------------
// Host-side orchestration
// ---------------------------------------------------------------------------
extern "C" void kernel_launch(void* const* d_in, const int* in_sizes, int n_in,
                              void* d_out, int out_size, void* d_ws, size_t ws_size,
                              hipStream_t stream) {
  (void)in_sizes; (void)n_in; (void)out_size; (void)ws_size;
  const float* txt    = (const float*)d_in[0];
  const float* vec    = (const float*)d_in[1];
  const float* mod_w  = (const float*)d_in[2];
  const float* mod_b  = (const float*)d_in[3];
  const float* q_w    = (const float*)d_in[4];
  const float* k_w    = (const float*)d_in[5];
  const float* v_w    = (const float*)d_in[6];
  const float* qn_w   = (const float*)d_in[7];
  const float* kn_w   = (const float*)d_in[8];
  const float* proj_w = (const float*)d_in[9];
  const float* fc1_w  = (const float*)d_in[10];
  const float* fc1_b  = (const float*)d_in[11];
  const float* fc2_w  = (const float*)d_in[12];
  const float* fc2_b  = (const float*)d_in[13];
  float* out = (float*)d_out;

  uint8_t* ws = (uint8_t*)d_ws;
  size_t off = 0;
  auto alloc = [&](size_t bytes) -> void* {
    void* p = ws + off;
    off += (bytes + 255) & ~(size_t)255;
    return p;
  };
  uint16_t* qwt   = (uint16_t*)alloc((size_t)CQ * CQ * 2);   // q_w^T  bf16
  uint16_t* kwt   = (uint16_t*)alloc((size_t)CQ * CQ * 2);
  uint16_t* vwt   = (uint16_t*)alloc((size_t)CQ * CQ * 2);
  uint16_t* pjt   = (uint16_t*)alloc((size_t)CQ * CQ * 2);
  uint16_t* f1t   = (uint16_t*)alloc((size_t)CM * CQ * 2);   // [CM, CQ]
  uint16_t* f2t   = (uint16_t*)alloc((size_t)CQ * CM * 2);   // [CQ, CM]
  float*    mod   = (float*)   alloc((size_t)BQ * C6 * 4);
  uint16_t* xm    = (uint16_t*)alloc((size_t)MQ * CQ * 2);   // LN1-modulated; reused for x2
  uint16_t* qb    = (uint16_t*)alloc((size_t)MQ * CQ * 2);   // rms-normed Q
  uint16_t* kb2   = (uint16_t*)alloc((size_t)MQ * CQ * 2);   // rms-normed K
  uint16_t* vT    = (uint16_t*)alloc((size_t)MQ * CQ * 2);   // V transposed: [BH*128][L]
  uint16_t* attnb = (uint16_t*)alloc((size_t)MQ * CQ * 2);   // attention output
  float*    tmid  = (float*)   alloc((size_t)MQ * CQ * 4);   // txt after first residual
  uint16_t* hb    = (uint16_t*)alloc((size_t)MQ * CM * 2);   // gelu(fc1) hidden
  uint16_t* x2    = xm;                                      // reuse: xm dead after QKV

  const dim3 tb(32, 8);
  transpose_f32_bf16<<<dim3(CQ / 32, CQ / 32), tb, 0, stream>>>(q_w, qwt, CQ, CQ);
  transpose_f32_bf16<<<dim3(CQ / 32, CQ / 32), tb, 0, stream>>>(k_w, kwt, CQ, CQ);
  transpose_f32_bf16<<<dim3(CQ / 32, CQ / 32), tb, 0, stream>>>(v_w, vwt, CQ, CQ);
  transpose_f32_bf16<<<dim3(CQ / 32, CQ / 32), tb, 0, stream>>>(proj_w, pjt, CQ, CQ);
  transpose_f32_bf16<<<dim3(CM / 32, CQ / 32), tb, 0, stream>>>(fc1_w, f1t, CQ, CM);
  transpose_f32_bf16<<<dim3(CQ / 32, CM / 32), tb, 0, stream>>>(fc2_w, f2t, CM, CQ);

  mod_gemv<<<dim3(C6 / 256, BQ), 256, 0, stream>>>(vec, mod_w, mod_b, mod);
  ln_modulate<<<MQ, 256, 0, stream>>>(txt, mod, 0 * CQ, 1 * CQ, xm);

  gemm_wmma<EPI_RMS><<<dim3(CQ / 128, MQ / 128), 256, 0, stream>>>(
      xm, qwt, MQ, CQ, CQ, qb, nullptr, nullptr, nullptr, 0, qn_w, LQ, HQ);
  gemm_wmma<EPI_RMS><<<dim3(CQ / 128, MQ / 128), 256, 0, stream>>>(
      xm, kwt, MQ, CQ, CQ, kb2, nullptr, nullptr, nullptr, 0, kn_w, LQ, HQ);
  gemm_wmma<EPI_VT><<<dim3(CQ / 128, MQ / 128), 256, 0, stream>>>(
      xm, vwt, MQ, CQ, CQ, vT, nullptr, nullptr, nullptr, 0, nullptr, LQ, HQ);

  attn_flash<<<dim3(BQ * HQ, LQ / 128), 256, 0, stream>>>(qb, kb2, vT, attnb);

  gemm_wmma<EPI_PROJ><<<dim3(CQ / 128, MQ / 128), 256, 0, stream>>>(
      attnb, pjt, MQ, CQ, CQ, tmid, nullptr, txt, mod + 2 * CQ, C6, nullptr, LQ, HQ);

  ln_modulate<<<MQ, 256, 0, stream>>>(tmid, mod, 3 * CQ, 4 * CQ, x2);

  gemm_wmma<EPI_GELU><<<dim3(CM / 128, MQ / 128), 256, 0, stream>>>(
      x2, f1t, MQ, CM, CQ, hb, fc1_b, nullptr, nullptr, 0, nullptr, LQ, HQ);
  gemm_wmma<EPI_FINAL><<<dim3(CQ / 128, MQ / 128), 256, 0, stream>>>(
      hb, f2t, MQ, CQ, CM, out, fc2_b, tmid, mod + 5 * CQ, C6, nullptr, LQ, HQ);
}
